// MultiScaleAttention_64424509440483
// MI455X (gfx1250) — matrix operand order repaired and checked by reading, not compiled, not run on hardware
//
#include <hip/hip_runtime.h>
#include <math.h>

typedef __attribute__((ext_vector_type(16))) _Float16 v16h;
typedef __attribute__((ext_vector_type(8)))  float    v8f;
typedef __attribute__((ext_vector_type(2)))  _Float16 h2;

#define S_LEN   4096
#define DIM     512
#define NHEAD   8
#define HDIM    64
#define NG      16
#define HALFWIN 32
#define SCALE   0.125f
#define LN_EPS  1e-5f

__device__ __forceinline__ v8f wmma_f16(v16h a, v16h b, v8f c) {
  // D = A(16x32 f16) * B(32x16 f16) + C(16x16 f32)
  return __builtin_amdgcn_wmma_f32_16x16x32_f16(false, a, false, b, (short)0, c,
                                                false, false);
}

// A tile (16x32 f16) in contiguous k order (k = 16*g + s, same as B); xp = &X[row(M=r) , kb]
__device__ __forceinline__ v16h load_16x32_A(const _Float16* xp, int g) {
  v16h a;
#pragma unroll
  for (int i = 0; i < 8; ++i) {
    int k0 = (g ? 16 : 0) + i * 2;
    h2 t = *(const h2*)(xp + k0);
    a[2*i] = t.x; a[2*i+1] = t.y;
  }
  return a;
}
// B tile (32x16 f16); wp = &W[col(N=r), kb + (g?16:0)]
__device__ __forceinline__ v16h load_32x16_B(const _Float16* wp) {
  v16h b;
#pragma unroll
  for (int j = 0; j < 8; ++j) {
    h2 t = *(const h2*)(wp + j*2);
    b[2*j] = t.x; b[2*j+1] = t.y;
  }
  return b;
}

// ---------------------------------------------------------------- f32 -> f16
__global__ void k_cvt_f16(const float* __restrict__ src, _Float16* __restrict__ dst, int n) {
  int i = blockIdx.x * 256 + threadIdx.x;
  if (i < n) dst[i] = (_Float16)src[i];
}

// ------------------------------------------------------ WMMA GEMM  Y = X*W^T + b
// X: [M,K] f16 row-major.  W: [N,K] f16 row-major (so W^T is KxN).
// One wave computes a 16(M) x 64(N) tile; K pipeline is double-buffered in
// 64-deep steps with the final step peeled.  Requires K % 64 == 0 and K >= 64.
// grid = (M/16, N/64), block = 32.
__global__ void k_gemm_wmma(const _Float16* __restrict__ X, const _Float16* __restrict__ W,
                            const float* __restrict__ bias,
                            float* __restrict__ Yf, _Float16* __restrict__ Yh,
                            int N, int K) {
  const int lane = threadIdx.x;
  const int g = lane >> 4, r = lane & 15;
  const int mbase = blockIdx.x * 16;
  const int nbase = blockIdx.y * 64;
  const int goff = g ? 16 : 0;

  const _Float16* xr = X + (size_t)(mbase + r) * K;
  const _Float16* w0 = W + (size_t)(nbase +  0 + r) * K + goff;
  const _Float16* w1 = W + (size_t)(nbase + 16 + r) * K + goff;
  const _Float16* w2 = W + (size_t)(nbase + 32 + r) * K + goff;
  const _Float16* w3 = W + (size_t)(nbase + 48 + r) * K + goff;

  v8f zero = {0,0,0,0,0,0,0,0};
  v8f acc0 = zero, acc1 = zero, acc2 = zero, acc3 = zero;

  // prologue: buffer A covers chunk kb=0
  v16h aA  = load_16x32_A(xr, g);
  v16h bA0 = load_32x16_B(w0);
  v16h bA1 = load_32x16_B(w1);
  v16h bA2 = load_32x16_B(w2);
  v16h bA3 = load_32x16_B(w3);

  int kb = 0;
  for (; kb < K - 64; kb += 64) {
    // buffer B covers chunk kb+32 (loads overlap WMMAs on buffer A)
    v16h aB  = load_16x32_A(xr + kb + 32, g);
    v16h bB0 = load_32x16_B(w0 + kb + 32);
    v16h bB1 = load_32x16_B(w1 + kb + 32);
    v16h bB2 = load_32x16_B(w2 + kb + 32);
    v16h bB3 = load_32x16_B(w3 + kb + 32);
    acc0 = wmma_f16(aA, bA0, acc0);
    acc1 = wmma_f16(aA, bA1, acc1);
    acc2 = wmma_f16(aA, bA2, acc2);
    acc3 = wmma_f16(aA, bA3, acc3);
    // buffer A reloads for chunk kb+64 (overlap WMMAs on buffer B)
    aA  = load_16x32_A(xr + kb + 64, g);
    bA0 = load_32x16_B(w0 + kb + 64);
    bA1 = load_32x16_B(w1 + kb + 64);
    bA2 = load_32x16_B(w2 + kb + 64);
    bA3 = load_32x16_B(w3 + kb + 64);
    acc0 = wmma_f16(aB, bB0, acc0);
    acc1 = wmma_f16(aB, bB1, acc1);
    acc2 = wmma_f16(aB, bB2, acc2);
    acc3 = wmma_f16(aB, bB3, acc3);
  }
  {
    // peeled final 64-deep step (kb == K-64): no further prefetch
    v16h aB  = load_16x32_A(xr + kb + 32, g);
    v16h bB0 = load_32x16_B(w0 + kb + 32);
    v16h bB1 = load_32x16_B(w1 + kb + 32);
    v16h bB2 = load_32x16_B(w2 + kb + 32);
    v16h bB3 = load_32x16_B(w3 + kb + 32);
    acc0 = wmma_f16(aA, bA0, acc0);
    acc1 = wmma_f16(aA, bA1, acc1);
    acc2 = wmma_f16(aA, bA2, acc2);
    acc3 = wmma_f16(aA, bA3, acc3);
    acc0 = wmma_f16(aB, bB0, acc0);
    acc1 = wmma_f16(aB, bB1, acc1);
    acc2 = wmma_f16(aB, bB2, acc2);
    acc3 = wmma_f16(aB, bB3, acc3);
  }

  v8f accv[4] = {acc0, acc1, acc2, acc3};
#pragma unroll
  for (int nt = 0; nt < 4; ++nt)
#pragma unroll
    for (int i = 0; i < 8; ++i) {
      int row = mbase + i + 8 * g;         // C layout: VGPR i, lane-half g
      int col = nbase + nt * 16 + r;
      float v = accv[nt][i] + (bias ? bias[col] : 0.f);
      if (Yf) Yf[(size_t)row * N + col] = v;
      if (Yh) Yh[(size_t)row * N + col] = (_Float16)v;
    }
}

// ------------------------------------------- banded local attention (flash-style)
// grid = (S/16, H), block = 32 (one wave).  Q,K,V: [S,DIM] f16.  Out: [S,DIM] f32.
__global__ void k_local_attn(const _Float16* __restrict__ Q, const _Float16* __restrict__ Kp,
                             const _Float16* __restrict__ V, float* __restrict__ Out) {
  __shared__ _Float16 pbuf[16 * 16];
  const int lane = threadIdx.x;
  const int g = lane >> 4, r = lane & 15;
  const int qbase = blockIdx.x * 16;
  const int hbase = blockIdx.y * HDIM;

  // Q A-tiles for feature chunks [0,32) and [32,64)
  const _Float16* qp = Q + (size_t)(qbase + r) * DIM + hbase;
  v16h aq0 = load_16x32_A(qp, g);
  v16h aq1 = load_16x32_A(qp + 32, g);

  v8f zero = {0,0,0,0,0,0,0,0};
  float mrow[8], lrow[8];
  v8f acc[4];
#pragma unroll
  for (int i = 0; i < 8; ++i) { mrow[i] = -1e30f; lrow[i] = 0.f; }
#pragma unroll
  for (int nt = 0; nt < 4; ++nt) acc[nt] = zero;

  for (int kt = -2; kt <= 2; ++kt) {            // 5 key tiles cover the +/-32 band
    int kbase = qbase + kt * 16;
    if (kbase < 0 || kbase >= S_LEN) continue;  // uniform across wave

    // scores (16x16) = Q(16x64) * K^T(64x16)
    v8f sc = zero;
    {
      const _Float16* kp = Kp + (size_t)(kbase + r) * DIM + hbase + (g ? 16 : 0);
      v16h b0 = load_32x16_B(kp);
      v16h b1 = load_32x16_B(kp + 32);
      sc = wmma_f16(aq0, b0, sc);
      sc = wmma_f16(aq1, b1, sc);
    }

    __syncthreads();  // pbuf reuse fence (prev iteration's reads done)
#pragma unroll
    for (int i = 0; i < 8; ++i) {
      int qpos = qbase + i + 8 * g;
      int kpos = kbase + r;
      int dlt = qpos - kpos; if (dlt < 0) dlt = -dlt;
      float si = sc[i] * SCALE;
      bool masked = (dlt > HALFWIN);
      if (masked) si = -1e30f;
      // row max over the 16 lanes holding this row (N dimension)
      float rm = si;
      rm = fmaxf(rm, __shfl_xor(rm, 1, 32));
      rm = fmaxf(rm, __shfl_xor(rm, 2, 32));
      rm = fmaxf(rm, __shfl_xor(rm, 4, 32));
      rm = fmaxf(rm, __shfl_xor(rm, 8, 32));
      float mnew = fmaxf(mrow[i], rm);
      float fac, p;
      if (mnew <= -1e29f) { fac = 1.f; p = 0.f; }   // fully-masked rows so far
      else {
        fac = __expf(mrow[i] - mnew);
        p = masked ? 0.f : __expf(si - mnew);
      }
      float rs = p;
      rs += __shfl_xor(rs, 1, 32);
      rs += __shfl_xor(rs, 2, 32);
      rs += __shfl_xor(rs, 4, 32);
      rs += __shfl_xor(rs, 8, 32);
      lrow[i] = lrow[i] * fac + rs;
      mrow[i] = mnew;
#pragma unroll
      for (int nt = 0; nt < 4; ++nt) acc[nt][i] *= fac;
      pbuf[(i + 8 * g) * 16 + r] = (_Float16)p;   // C layout -> LDS (M,N)
    }
    __syncthreads();

    // P re-read in A layout (16x16 valid, K padded to 32 with zeros)
    v16h ap;
#pragma unroll
    for (int i = 0; i < 8; ++i) {
      int k0 = (g ? 16 : 0) + i * 2;
      _Float16 e0 = (_Float16)0.f, e1 = (_Float16)0.f;
      if (k0 < 16) { e0 = pbuf[r * 16 + k0]; e1 = pbuf[r * 16 + k0 + 1]; }
      ap[2*i] = e0; ap[2*i+1] = e1;
    }
#pragma unroll
    for (int nt = 0; nt < 4; ++nt) {
      v16h bv;   // V tile as B(32x16): K rows 0..15 = keys, 16..31 = zero pad
#pragma unroll
      for (int j = 0; j < 8; ++j) {
        int k = (g ? 16 : 0) + j * 2;
        _Float16 e0 = (_Float16)0.f, e1 = (_Float16)0.f;
        if (k < 16) {
          e0 = V[(size_t)(kbase + k)     * DIM + hbase + nt * 16 + r];
          e1 = V[(size_t)(kbase + k + 1) * DIM + hbase + nt * 16 + r];
        }
        bv[2*j] = e0; bv[2*j+1] = e1;
      }
      acc[nt] = wmma_f16(ap, bv, acc[nt]);
    }
  }

#pragma unroll
  for (int i = 0; i < 8; ++i) {
    float inv = 1.f / lrow[i];   // diagonal always unmasked -> lrow > 0
    int row = qbase + i + 8 * g;
#pragma unroll
    for (int nt = 0; nt < 4; ++nt)
      Out[(size_t)row * DIM + hbase + nt * 16 + r] = acc[nt][i] * inv;
  }
}

// --------------------------------------------- tiny f32 linear  Y = A*W^T + b
__global__ void k_lin_f32(const float* __restrict__ A, const float* __restrict__ W,
                          const float* __restrict__ bias, float* __restrict__ Y,
                          int R, int N, int K) {
  int idx = blockIdx.x * 256 + threadIdx.x;
  if (idx >= R * N) return;
  int r = idx / N, n = idx - r * N;
  float s = bias[n];
  const float* a = A + (size_t)r * K;
  const float* w = W + (size_t)n * K;
  for (int k = 0; k < K; ++k) s += a[k] * w[k];
  Y[idx] = s;
}

// -------------------- global attention stage 1: summaries attend to sequence
// one workgroup (256 threads) per (h,g).  qg f32 [G,DIM]; kx,vx f16 [S,DIM].
__global__ void k_gattn1(const float* __restrict__ qg, const _Float16* __restrict__ kx,
                         const _Float16* __restrict__ vx, float* __restrict__ summ) {
  __shared__ float sc[S_LEN];
  __shared__ float qv[HDIM];
  __shared__ float red[256];
  const int h = blockIdx.x >> 4, g = blockIdx.x & 15;
  const int tid = threadIdx.x;

  if (tid < HDIM) qv[tid] = qg[g * DIM + h * HDIM + tid];
  __syncthreads();

  float lmax = -1e30f;
  for (int j = tid; j < S_LEN; j += 256) {
    const _Float16* kp = kx + (size_t)j * DIM + h * HDIM;
    float s = 0.f;
    for (int d = 0; d < HDIM; ++d) s += qv[d] * (float)kp[d];
    s *= SCALE;
    sc[j] = s;
    lmax = fmaxf(lmax, s);
  }
  red[tid] = lmax; __syncthreads();
  for (int st = 128; st > 0; st >>= 1) { if (tid < st) red[tid] = fmaxf(red[tid], red[tid + st]); __syncthreads(); }
  float m = red[0]; __syncthreads();

  float lsum = 0.f;
  for (int j = tid; j < S_LEN; j += 256) { float p = __expf(sc[j] - m); sc[j] = p; lsum += p; }
  red[tid] = lsum; __syncthreads();
  for (int st = 128; st > 0; st >>= 1) { if (tid < st) red[tid] += red[tid + st]; __syncthreads(); }
  float inv = 1.f / red[0]; __syncthreads();

  const int d = tid & 63, chunk = tid >> 6;   // 4 chunks of 1024 keys
  float part = 0.f;
  for (int j = chunk * 1024; j < (chunk + 1) * 1024; ++j)
    part += sc[j] * (float)vx[(size_t)j * DIM + h * HDIM + d];
  red[tid] = part; __syncthreads();
  if (tid < HDIM)
    summ[g * DIM + h * HDIM + d] =
        (red[tid] + red[tid + 64] + red[tid + 128] + red[tid + 192]) * inv;
}

// ------------- global attention stage 2: sequence attends to 16 summary tokens
__global__ void k_gattn2(const _Float16* __restrict__ QS, const float* __restrict__ KG,
                         const float* __restrict__ VG, float* __restrict__ Glob) {
  int idx = blockIdx.x * 256 + threadIdx.x;
  if (idx >= S_LEN * NHEAD) return;
  int i = idx >> 3, h = idx & 7;
  const _Float16* qp = QS + (size_t)i * DIM + h * HDIM;
  float p[NG]; float mx = -1e30f;
  for (int g = 0; g < NG; ++g) {
    const float* kp = KG + g * DIM + h * HDIM;
    float s = 0.f;
    for (int d = 0; d < HDIM; ++d) s += (float)qp[d] * kp[d];
    s *= SCALE;
    p[g] = s; mx = fmaxf(mx, s);
  }
  float sum = 0.f;
  for (int g = 0; g < NG; ++g) { p[g] = __expf(p[g] - mx); sum += p[g]; }
  float inv = 1.f / sum;
  float* op = Glob + (size_t)i * DIM + h * HDIM;
  for (int d = 0; d < HDIM; ++d) {
    float a = 0.f;
    for (int g = 0; g < NG; ++g) a += p[g] * VG[g * DIM + h * HDIM + d];
    op[d] = a * inv;
  }
}

// ----------------- LayerNorm over DIM, write f16 into concat buffer [S, 2*DIM]
__global__ void k_ln(const float* __restrict__ src, const float* __restrict__ gamma,
                     const float* __restrict__ beta, _Float16* __restrict__ dst,
                     int col_off) {
  __shared__ float red[256];
  const int row = blockIdx.x, tid = threadIdx.x;
  float a0 = src[(size_t)row * DIM + tid];
  float a1 = src[(size_t)row * DIM + 256 + tid];
  red[tid] = a0 + a1; __syncthreads();
  for (int st = 128; st > 0; st >>= 1) { if (tid < st) red[tid] += red[tid + st]; __syncthreads(); }
  float mean = red[0] * (1.f / (float)DIM); __syncthreads();
  float d0 = a0 - mean, d1 = a1 - mean;
  red[tid] = d0 * d0 + d1 * d1; __syncthreads();
  for (int st = 128; st > 0; st >>= 1) { if (tid < st) red[tid] += red[tid + st]; __syncthreads(); }
  float rstd = rsqrtf(red[0] * (1.f / (float)DIM) + LN_EPS);
  _Float16* dp = dst + (size_t)row * (2 * DIM) + col_off;
  dp[tid]       = (_Float16)(d0 * rstd * gamma[tid]       + beta[tid]);
  dp[256 + tid] = (_Float16)(d1 * rstd * gamma[256 + tid] + beta[256 + tid]);
}

// =============================================================================
extern "C" void kernel_launch(void* const* d_in, const int* in_sizes, int n_in,
                              void* d_out, int out_size, void* d_ws, size_t ws_size,
                              hipStream_t stream) {
  (void)in_sizes; (void)n_in; (void)out_size; (void)ws_size;
  const float* x      = (const float*)d_in[0];
  const float* lq_w   = (const float*)d_in[1];
  const float* lq_b   = (const float*)d_in[2];
  const float* lk_w   = (const float*)d_in[3];
  const float* lk_b   = (const float*)d_in[4];
  const float* lv_w   = (const float*)d_in[5];
  const float* lv_b   = (const float*)d_in[6];
  const float* gq_w   = (const float*)d_in[7];
  const float* gq_b   = (const float*)d_in[8];
  const float* gk_w   = (const float*)d_in[9];
  const float* gk_b   = (const float*)d_in[10];
  const float* gv_w   = (const float*)d_in[11];
  const float* gv_b   = (const float*)d_in[12];
  const float* gtok   = (const float*)d_in[13];
  const float* out_w  = (const float*)d_in[14];
  const float* out_b  = (const float*)d_in[15];
  const float* ln_l_w = (const float*)d_in[16];
  const float* ln_l_b = (const float*)d_in[17];
  const float* ln_g_w = (const float*)d_in[18];
  const float* ln_g_b = (const float*)d_in[19];

  char* p = (char*)d_ws;
  auto alloc = [&](size_t bytes) { char* q = p; p += (bytes + 255) & ~(size_t)255; return q; };
  const size_t SD2 = (size_t)S_LEN * DIM * sizeof(_Float16);
  const size_t SD4 = (size_t)S_LEN * DIM * sizeof(float);
  const size_t WW2 = (size_t)DIM * DIM * sizeof(_Float16);

  _Float16* xh     = (_Float16*)alloc(SD2);
  _Float16* qh     = (_Float16*)alloc(SD2);
  _Float16* kh     = (_Float16*)alloc(SD2);
  _Float16* vh     = (_Float16*)alloc(SD2);
  _Float16* kxh    = (_Float16*)alloc(SD2);
  _Float16* vxh    = (_Float16*)alloc(SD2);
  _Float16* qsh    = (_Float16*)alloc(SD2);
  _Float16* lqh    = (_Float16*)alloc(WW2);
  _Float16* lkh    = (_Float16*)alloc(WW2);
  _Float16* lvh    = (_Float16*)alloc(WW2);
  _Float16* gqh    = (_Float16*)alloc(WW2);
  _Float16* gkh    = (_Float16*)alloc(WW2);
  _Float16* gvh    = (_Float16*)alloc(WW2);
  _Float16* outh   = (_Float16*)alloc((size_t)DIM * 2 * DIM * sizeof(_Float16));
  float*    qgf    = (float*)alloc((size_t)NG * DIM * sizeof(float));
  float*    summf  = (float*)alloc((size_t)NG * DIM * sizeof(float));
  float*    kgf    = (float*)alloc((size_t)NG * DIM * sizeof(float));
  float*    vgf    = (float*)alloc((size_t)NG * DIM * sizeof(float));
  float*    localf = (float*)alloc(SD4);
  float*    globf  = (float*)alloc(SD4);
  _Float16* conch  = (_Float16*)alloc((size_t)S_LEN * 2 * DIM * sizeof(_Float16));

  // ---- phase 0: f32 -> f16 conversions
  const int nx = S_LEN * DIM, nw = DIM * DIM, no = DIM * 2 * DIM;
  k_cvt_f16<<<(nx + 255) / 256, 256, 0, stream>>>(x, xh, nx);
  k_cvt_f16<<<(nw + 255) / 256, 256, 0, stream>>>(lq_w, lqh, nw);
  k_cvt_f16<<<(nw + 255) / 256, 256, 0, stream>>>(lk_w, lkh, nw);
  k_cvt_f16<<<(nw + 255) / 256, 256, 0, stream>>>(lv_w, lvh, nw);
  k_cvt_f16<<<(nw + 255) / 256, 256, 0, stream>>>(gq_w, gqh, nw);
  k_cvt_f16<<<(nw + 255) / 256, 256, 0, stream>>>(gk_w, gkh, nw);
  k_cvt_f16<<<(nw + 255) / 256, 256, 0, stream>>>(gv_w, gvh, nw);
  k_cvt_f16<<<(no + 255) / 256, 256, 0, stream>>>(out_w, outh, no);

  // ---- phase 1: six WMMA projections of x (16M x 64N per wave, pipelined K)
  dim3 gproj(S_LEN / 16, DIM / 64);
  k_gemm_wmma<<<gproj, 32, 0, stream>>>(xh, lqh, lq_b, nullptr, qh,  DIM, DIM);
  k_gemm_wmma<<<gproj, 32, 0, stream>>>(xh, lkh, lk_b, nullptr, kh,  DIM, DIM);
  k_gemm_wmma<<<gproj, 32, 0, stream>>>(xh, lvh, lv_b, nullptr, vh,  DIM, DIM);
  k_gemm_wmma<<<gproj, 32, 0, stream>>>(xh, gkh, gk_b, nullptr, kxh, DIM, DIM);
  k_gemm_wmma<<<gproj, 32, 0, stream>>>(xh, gvh, gv_b, nullptr, vxh, DIM, DIM);
  k_gemm_wmma<<<gproj, 32, 0, stream>>>(xh, gqh, gq_b, nullptr, qsh, DIM, DIM);

  // ---- phase 2: qg = gtok @ gq_w^T + b   (tiny, f32)
  k_lin_f32<<<(NG * DIM + 255) / 256, 256, 0, stream>>>(gtok, gq_w, gq_b, qgf, NG, DIM, DIM);

  // ---- phase 3: summaries attend to sequence -> summ [G,DIM]
  k_gattn1<<<NHEAD * NG, 256, 0, stream>>>(qgf, kxh, vxh, summf);

  // ---- phase 4: kg/vg = summ @ gk_w^T / gv_w^T  (tiny, f32)
  k_lin_f32<<<(NG * DIM + 255) / 256, 256, 0, stream>>>(summf, gk_w, gk_b, kgf, NG, DIM, DIM);
  k_lin_f32<<<(NG * DIM + 255) / 256, 256, 0, stream>>>(summf, gv_w, gv_b, vgf, NG, DIM, DIM);

  // ---- phase 5: sequence attends to summaries -> glob (pre-LN)
  k_gattn2<<<(S_LEN * NHEAD + 255) / 256, 256, 0, stream>>>(qsh, kgf, vgf, globf);

  // ---- phase 6: banded local attention (WMMA flash) -> local (pre-LN)
  dim3 gloc(S_LEN / 16, NHEAD);
  k_local_attn<<<gloc, 32, 0, stream>>>(qh, kh, vh, localf);

  // ---- phase 7: LayerNorms into concat [S, 2*DIM] f16
  k_ln<<<S_LEN, 256, 0, stream>>>(localf, ln_l_w, ln_l_b, conch, 0);
  k_ln<<<S_LEN, 256, 0, stream>>>(globf,  ln_g_w, ln_g_b, conch, DIM);

  // ---- phase 8: out = concat @ out_w^T + out_b  (f32 output)
  dim3 gout(S_LEN / 16, DIM / 64);
  k_gemm_wmma<<<gout, 32, 0, stream>>>(conch, outh, out_b, (float*)d_out, nullptr, DIM, 2 * DIM);
}
